// Model_34394098106691
// MI455X (gfx1250) — matrix operand order repaired and checked by reading, not compiled, not run on hardware
//
#include <hip/hip_runtime.h>

// ---------------------------------------------------------------------------
// MI455X (gfx1250). Heavy math = bf16 WMMA, f32 accumulate
// (v_wmma_f32_16x16x32_bf16). FFTs are DFT-matrix GEMMs. All B operands are
// pre-packed TRANSPOSED ([M x K] row-major) so both A and B tiles stage into
// LDS as contiguous 16B chunks via CDNA5 async global->LDS copies
// (global_load_async_to_lds_b128, ASYNCcnt) with double-buffered LDS tiles.
// ---------------------------------------------------------------------------

#define USE_ASYNC 1

typedef unsigned short u16;
typedef __attribute__((ext_vector_type(16))) __bf16 v16bf;
typedef __attribute__((ext_vector_type(8)))  float  v8f;

#define B_   64
#define L_   512
#define P_   512
#define C_   256
#define NB_  3
#define E_   8
#define F_   257
#define PF_  513
#define N_   (B_*C_)      // 16384 rows
#define PADF 320          // 257 padded to 5*64
#define PADP 576          // 513 padded to 9*64
#define LOUT 1024

#define OUT_TOTAL ((size_t)B_*P_*C_)   // 8388608
#define OUT_BOUNDS (OUT_TOTAL)
#define OUT_WEIGHT (OUT_TOTAL + 9)

// ------------------------------ helpers ------------------------------------

__device__ __forceinline__ u16 f2bf(float x) {
  unsigned u = __builtin_bit_cast(unsigned, x);
  unsigned r = (u + 0x7FFFu + ((u >> 16) & 1u)) >> 16;   // RNE
  return (u16)r;
}
__device__ __forceinline__ float bf2f(u16 h) {
  unsigned u = ((unsigned)h) << 16;
  return __builtin_bit_cast(float, u);
}
__device__ __forceinline__ v8f vzero() {
  v8f z = {0.f,0.f,0.f,0.f,0.f,0.f,0.f,0.f};
  return z;
}

union FragU { v16bf v; uint4 q[2]; };

// A fragment: 16x32 bf16 (ISA 7.12.2): lanes0-15 M=0..15 {K0..7,K16..23},
// lanes16-31 {K8..15,K24..31}. Two b128 LDS reads.
__device__ __forceinline__ v16bf ldsA(const u16* sm, int row, int hf) {
  FragU f;
  const u16* p = sm + row * 32;
  f.q[0] = *(const uint4*)(p + hf * 8);
  f.q[1] = *(const uint4*)(p + 16 + hf * 8);
  return f.v;
}
// B fragment from LDS tile bt[col][k]:
// lanes0-15 N=col {K0..7,K16..23}, lanes16-31 {K8..15,K24..31}.
__device__ __forceinline__ v16bf ldsB(const u16* sm, int col, int hf) {
  FragU f;
  const u16* p = sm + col * 32;
  f.q[0] = *(const uint4*)(p + hf * 8);
  f.q[1] = *(const uint4*)(p + 16 + hf * 8);
  return f.v;
}
__device__ __forceinline__ v8f wmma_bf16(v16bf a, v16bf b, v8f c) {
  return __builtin_amdgcn_wmma_f32_16x16x32_bf16(false, a, false, b,
                                                 (short)0, c, false, false);
}

// 16-byte global -> LDS copy. Async on CDNA5 (tracked by ASYNCcnt).
__device__ __forceinline__ void copy16(const u16* g, u16* sm) {
#if USE_ASYNC
  unsigned ldsoff = (unsigned)(size_t)sm;   // low 32 bits = LDS offset
  unsigned long long ga = (unsigned long long)(size_t)g;
  asm volatile("global_load_async_to_lds_b128 %0, %1, off"
               :: "v"(ldsoff), "v"(ga) : "memory");
#else
  *(uint4*)sm = *(const uint4*)g;
#endif
}
__device__ __forceinline__ void async_wait_all() {
#if USE_ASYNC
  asm volatile("s_wait_asynccnt 0x0" ::: "memory");
#endif
}

// Stage a 64x32 bf16 tile (rows row0.., k cols k0..k0+31) from a row-major
// matrix with leading dim ld into LDS tile sm[64][32]. 128 threads, 2x16B.
__device__ __forceinline__ void stage(u16* sm, const u16* g, int ld,
                                      int row0, int k0, int tid) {
#pragma unroll
  for (int j = 0; j < 2; ++j) {
    int chunk = tid * 2 + j;            // 0..255
    int r = chunk >> 2;                 // row 0..63
    int o = (chunk & 3) * 8;            // 0,8,16,24 halves
    copy16(g + (size_t)(row0 + r) * ld + k0 + o, sm + r * 32 + o);
  }
}

// ------------------------------ basis generation ---------------------------
// All bases stored TRANSPOSED: [M x K] row-major (M = output cols of GEMM,
// K = reduction dim), so staging is contiguous.

__global__ void k_basis_rfft(u16* bc, u16* bs) {   // B^T: [PADF x 512]
  int i = blockIdx.x * 256 + threadIdx.x;
  if (i >= PADF * 512) return;
  int f = i / 512, l = i % 512;
  float c = 0.f, s = 0.f;
  if (f < F_) {
    int m = (f * l) & (L_ - 1);
    float th = (float)m * (6.283185307179586f / (float)L_);
    float sn, cs; __sincosf(th, &sn, &cs);
    c = cs; s = -sn;
  }
  bc[i] = f2bf(c); bs[i] = f2bf(s);
}
__global__ void k_basis_ir512(u16* bc, u16* bs) {  // B^T: [512 x PADF]
  int i = blockIdx.x * 256 + threadIdx.x;
  if (i >= 512 * PADF) return;
  int l = i / PADF, f = i % PADF;
  float c = 0.f, s = 0.f;
  if (f < F_) {
    int m = (f * l) & (L_ - 1);
    float th = (float)m * (6.283185307179586f / (float)L_);
    float sn, cs; __sincosf(th, &sn, &cs);
    float w = (f == 0 || f == F_ - 1) ? 1.f : 2.f;
    c = w * cs * (1.f / (float)L_);
    s = -w * sn * (1.f / (float)L_);
  }
  bc[i] = f2bf(c); bs[i] = f2bf(s);
}
__global__ void k_basis_ir1024(u16* bc, u16* bs) { // B^T: [1024 x PADP]
  int i = blockIdx.x * 256 + threadIdx.x;
  if (i >= LOUT * PADP) return;
  int t = i / PADP, f = i % PADP;
  float c = 0.f, s = 0.f;
  if (f < PF_) {
    int m = (f * t) & (LOUT - 1);
    float th = (float)m * (6.283185307179586f / (float)LOUT);
    float sn, cs; __sincosf(th, &sn, &cs);
    float w = (f == 0 || f == PF_ - 1) ? 1.f : 2.f;
    c = w * cs * (1.f / (float)LOUT);
    s = -w * sn * (1.f / (float)LOUT);
  }
  bc[i] = f2bf(c); bs[i] = f2bf(s);
}

// ------------------------------ weight packing -----------------------------
// GEMM1: up[n,p] = sum_f spec[n,f]*w1[p,f]  -> B^T[p][f] = w1[p][f] (direct!)
__global__ void k_pack_w1(const float* w1, u16* oR, u16* oI, u16* oN) {
  int i = blockIdx.x * 256 + threadIdx.x;
  if (i >= NB_ * PADP * PADF) return;
  int blk = i / (PADP * PADF);
  int rem = i % (PADP * PADF);
  int p = rem / PADF, f = rem % PADF;
  float re = 0.f, im = 0.f;
  if (p < PF_ && f < F_) {
    size_t s = ((size_t)(blk * PF_ + p) * F_ + f) * 2;
    re = w1[s]; im = w1[s + 1];
  }
  oR[i] = f2bf(re); oI[i] = f2bf(im); oN[i] = f2bf(-im);
}
// GEMM2: up2[n,q] = sum_p up[n,p]*w2[q,p] -> B^T[q][p] = w2[q][p] (direct!)
__global__ void k_pack_w2(const float* w2, u16* oR, u16* oI, u16* oN) {
  int i = blockIdx.x * 256 + threadIdx.x;
  if (i >= NB_ * PADP * PADP) return;
  int blk = i / (PADP * PADP);
  int rem = i % (PADP * PADP);
  int q = rem / PADP, p = rem % PADP;
  float re = 0.f, im = 0.f;
  if (q < PF_ && p < PF_) {
    size_t s = ((size_t)(blk * PF_ + q) * PF_ + p) * 2;
    re = w2[s]; im = w2[s + 1];
  }
  oR[i] = f2bf(re); oI[i] = f2bf(im); oN[i] = f2bf(-im);
}

// ------------------------------ misc kernels -------------------------------

__global__ void k_zero(float* p, size_t n) {
  size_t i = (size_t)blockIdx.x * 256 + threadIdx.x;
  if (i < n) p[i] = 0.f;
}

// Per-row mean / std (ddof=1, +1e-5) + normalized bf16 rows.
__global__ __launch_bounds__(256) void k_stats(const float* src, int strided,
                                               u16* xn, float* mean, float* stdv) {
  int n = blockIdx.x, t = threadIdx.x;
  int b = n >> 8, c = n & 255;
  float v1, v2;
  if (strided) {
    v1 = src[((size_t)b * L_ + t) * C_ + c];
    v2 = src[((size_t)b * L_ + t + 256) * C_ + c];
  } else {
    v1 = src[(size_t)n * L_ + t];
    v2 = src[(size_t)n * L_ + t + 256];
  }
  __shared__ float sS[256], sQ[256];
  sS[t] = v1 + v2;
  sQ[t] = v1 * v1 + v2 * v2;
  __syncthreads();
  for (int s = 128; s > 0; s >>= 1) {
    if (t < s) { sS[t] += sS[t + s]; sQ[t] += sQ[t + s]; }
    __syncthreads();
  }
  float mu  = sS[0] * (1.f / (float)L_);
  float var = (sQ[0] - (float)L_ * mu * mu) * (1.f / (float)(L_ - 1)) + 1e-5f;
  float sd  = sqrtf(var);
  float inv = 1.f / sd;
  if (t == 0) { mean[n] = mu; stdv[n] = sd; }
  xn[(size_t)n * L_ + t]       = f2bf((v1 - mu) * inv);
  xn[(size_t)n * L_ + t + 256] = f2bf((v2 - mu) * inv);
}

__global__ void k_gate_absmean(const u16* sr, const u16* si, float* gi) {
  int idx = blockIdx.x * 256 + threadIdx.x;
  if (idx >= B_ * PADF) return;
  int b = idx / PADF, f = idx % PADF;
  float s = 0.f;
  if (f < F_) {
    const u16* pr = sr + (size_t)b * C_ * PADF + f;
    const u16* pi = si + (size_t)b * C_ * PADF + f;
    for (int c = 0; c < C_; ++c) {
      float re = bf2f(pr[(size_t)c * PADF]);
      float im = bf2f(pi[(size_t)c * PADF]);
      s += sqrtf(re * re + im * im);
    }
    s *= (1.f / (float)C_);
  }
  gi[idx] = s;
}

__global__ void k_gate_h(const float* gi, const float* gw1, const float* gb1,
                         float* h) {
  int idx = blockIdx.x * 256 + threadIdx.x;
  if (idx >= B_ * PADF) return;
  int b = idx / PADF, j = idx % PADF;
  float s = 0.f;
  if (j < F_) {
    const float* g = gi + (size_t)b * PADF;
    const float* w = gw1 + (size_t)j * F_;
    for (int f = 0; f < F_; ++f) s += g[f] * w[f];
    s = fmaxf(s + gb1[j], 0.f);
  }
  h[idx] = s;
}

__global__ void k_gate_scores(const float* h, const float* gw2, const float* gb2,
                              const float* bb, float* scores,
                              float* outBounds, float* outWeight, int* idxb) {
  int b = threadIdx.x;   // 64 threads
  if (b < B_) {
    float lg[E_];
    const float* hb = h + (size_t)b * PADF;
    float mx = -1e30f;
    for (int e = 0; e < E_; ++e) {
      float s = gb2[e];
      const float* w = gw2 + (size_t)e * F_;
      for (int j = 0; j < F_; ++j) s += hb[j] * w[j];
      lg[e] = s; mx = fmaxf(mx, s);
    }
    float den = 0.f;
    for (int e = 0; e < E_; ++e) { lg[e] = __expf(lg[e] - mx); den += lg[e]; }
    float inv = 1.f / den;
    for (int e = 0; e < E_; ++e) {
      float v = lg[e] * inv;
      scores[b * E_ + e] = v;
      outWeight[b * E_ + e] = v;
    }
  }
  if (b == 0) {
    float v[E_ - 1];
    for (int e = 0; e < E_ - 1; ++e) v[e] = 1.f / (1.f + __expf(-bb[e]));
    for (int a = 1; a < E_ - 1; ++a) {        // insertion sort (7 values)
      float key = v[a]; int c = a - 1;
      while (c >= 0 && v[c] > key) { v[c + 1] = v[c]; --c; }
      v[c + 1] = key;
    }
    float bd[E_ + 1];
    bd[0] = 0.f;
    for (int e = 0; e < E_ - 1; ++e) bd[e + 1] = v[e];
    bd[E_] = 1.f;
    for (int e = 0; e <= E_; ++e) {
      outBounds[e] = bd[e];
      idxb[e] = (e == E_) ? F_ : (int)(bd[e] * (float)F_);
    }
  }
}

__global__ void k_wbin(const float* scores, const int* idxb, float* wbin) {
  int idx = blockIdx.x * 256 + threadIdx.x;
  if (idx >= B_ * PADF) return;
  int b = idx / PADF, f = idx % PADF;
  float w = 0.f;
  if (f < F_)
    for (int e = 0; e < E_; ++e)
      if (f >= idxb[e] && f < idxb[e + 1]) w += scores[b * E_ + e];
  wbin[idx] = w;
}

__global__ void k_scale_fx(const u16* sr, const u16* si, const float* wbin,
                           u16* fr, u16* fi) {
  size_t idx = (size_t)blockIdx.x * 256 + threadIdx.x;
  if (idx >= (size_t)N_ * PADF) return;
  int f = (int)(idx % PADF);
  int n = (int)(idx / PADF);
  float w = wbin[(n >> 8) * PADF + f];
  fr[idx] = f2bf(bf2f(sr[idx]) * w);
  fi[idx] = f2bf(bf2f(si[idx]) * w);
}

// ------------------------------ WMMA GEMM kernels --------------------------
// 64x64 output tile, 128 threads (4 waves), K-step 32, double-buffered LDS
// filled by async global->LDS copies; all fragments loaded before WMMA chain.

// spec_{r,i}[N x PADF] = xn[N x 512] @ {cos,-sin};  B^T given [PADF x 512].
__global__ __launch_bounds__(128) void k_rfft_gemm(
    const u16* __restrict__ A, int lda,
    const u16* __restrict__ BcT, const u16* __restrict__ BsT, int ldb, int K,
    u16* __restrict__ outR, u16* __restrict__ outI, int ldo) {
  __shared__ u16 a_sm[2][2048], bc_sm[2][2048], bs_sm[2][2048];
  const int tid = threadIdx.x;
  const int row0 = blockIdx.y * 64, col0 = blockIdx.x * 64;
  const int wv = tid >> 5, ln = tid & 31, m16 = ln & 15, hf = ln >> 4;
  v8f accR[4], accI[4];
#pragma unroll
  for (int j = 0; j < 4; ++j) { accR[j] = vzero(); accI[j] = vzero(); }
  stage(a_sm[0], A, lda, row0, 0, tid);
  stage(bc_sm[0], BcT, ldb, col0, 0, tid);
  stage(bs_sm[0], BsT, ldb, col0, 0, tid);
  const int nk = K >> 5;
  for (int ki = 0; ki < nk; ++ki) {
    const int cur = ki & 1;
    async_wait_all();
    __syncthreads();
    if (ki + 1 < nk) {
      int k1 = (ki + 1) << 5;
      stage(a_sm[1 - cur], A, lda, row0, k1, tid);
      stage(bc_sm[1 - cur], BcT, ldb, col0, k1, tid);
      stage(bs_sm[1 - cur], BsT, ldb, col0, k1, tid);
    }
    v16bf af = ldsA(a_sm[cur], wv * 16 + m16, hf);
    v16bf bcf[4], bsf[4];
#pragma unroll
    for (int j = 0; j < 4; ++j) {
      bcf[j] = ldsB(bc_sm[cur], j * 16 + m16, hf);
      bsf[j] = ldsB(bs_sm[cur], j * 16 + m16, hf);
    }
#pragma unroll
    for (int j = 0; j < 4; ++j) {
      accR[j] = wmma_bf16(af, bcf[j], accR[j]);
      accI[j] = wmma_bf16(af, bsf[j], accI[j]);
    }
  }
#pragma unroll
  for (int j = 0; j < 4; ++j)
#pragma unroll
    for (int r = 0; r < 8; ++r) {
      int grow = row0 + wv * 16 + hf * 8 + r;
      int gcol = col0 + j * 16 + m16;
      outR[(size_t)grow * ldo + gcol] = f2bf(accR[j][r]);
      outI[(size_t)grow * ldo + gcol] = f2bf(accI[j][r]);
    }
}

// Complex GEMM: out = (Ar + iAi) @ (Br + iBi); BnT == -Bi prepacked.
__global__ __launch_bounds__(128) void k_cgemm(
    const u16* __restrict__ Ar, const u16* __restrict__ Ai, int lda, int K,
    const u16* __restrict__ BrT, const u16* __restrict__ BiT,
    const u16* __restrict__ BnT, int ldb,
    u16* __restrict__ outR, u16* __restrict__ outI, int ldo,
    const float* __restrict__ bias, int biasLim, int doRelu) {
  __shared__ u16 ar_sm[2][2048], ai_sm[2][2048];
  __shared__ u16 br_sm[2][2048], bi_sm[2][2048], bn_sm[2][2048];
  const int tid = threadIdx.x;
  const int row0 = blockIdx.y * 64, col0 = blockIdx.x * 64;
  const int wv = tid >> 5, ln = tid & 31, m16 = ln & 15, hf = ln >> 4;
  v8f accR[4], accI[4];
#pragma unroll
  for (int j = 0; j < 4; ++j) { accR[j] = vzero(); accI[j] = vzero(); }
  stage(ar_sm[0], Ar, lda, row0, 0, tid);
  stage(ai_sm[0], Ai, lda, row0, 0, tid);
  stage(br_sm[0], BrT, ldb, col0, 0, tid);
  stage(bi_sm[0], BiT, ldb, col0, 0, tid);
  stage(bn_sm[0], BnT, ldb, col0, 0, tid);
  const int nk = K >> 5;
  for (int ki = 0; ki < nk; ++ki) {
    const int cur = ki & 1;
    async_wait_all();
    __syncthreads();
    if (ki + 1 < nk) {
      int k1 = (ki + 1) << 5;
      stage(ar_sm[1 - cur], Ar, lda, row0, k1, tid);
      stage(ai_sm[1 - cur], Ai, lda, row0, k1, tid);
      stage(br_sm[1 - cur], BrT, ldb, col0, k1, tid);
      stage(bi_sm[1 - cur], BiT, ldb, col0, k1, tid);
      stage(bn_sm[1 - cur], BnT, ldb, col0, k1, tid);
    }
    v16bf arf = ldsA(ar_sm[cur], wv * 16 + m16, hf);
    v16bf aif = ldsA(ai_sm[cur], wv * 16 + m16, hf);
    v16bf brf[4], bif[4], bnf[4];
#pragma unroll
    for (int j = 0; j < 4; ++j) {
      brf[j] = ldsB(br_sm[cur], j * 16 + m16, hf);
      bif[j] = ldsB(bi_sm[cur], j * 16 + m16, hf);
      bnf[j] = ldsB(bn_sm[cur], j * 16 + m16, hf);
    }
#pragma unroll
    for (int j = 0; j < 4; ++j) {
      accR[j] = wmma_bf16(arf, brf[j], accR[j]);  // +Ar*Br
      accR[j] = wmma_bf16(aif, bnf[j], accR[j]);  // -Ai*Bi
      accI[j] = wmma_bf16(arf, bif[j], accI[j]);  // +Ar*Bi
      accI[j] = wmma_bf16(aif, brf[j], accI[j]);  // +Ai*Br
    }
  }
#pragma unroll
  for (int j = 0; j < 4; ++j)
#pragma unroll
    for (int r = 0; r < 8; ++r) {
      int grow = row0 + wv * 16 + hf * 8 + r;
      int gcol = col0 + j * 16 + m16;
      float re = accR[j][r], im = accI[j][r];
      if (gcol < biasLim) { re += bias[2 * gcol]; im += bias[2 * gcol + 1]; }
      if (doRelu) { re = fmaxf(re, 0.f); im = fmaxf(im, 0.f); }
      outR[(size_t)grow * ldo + gcol] = f2bf(re);
      outI[(size_t)grow * ldo + gcol] = f2bf(im);
    }
}

// irfft GEMM: acc = Ar@Bc + Ai@Bs (bases carry 2/N weights and -sin).
// mode 0 (MoE):  residual[n,t] = acc*std + mean
// mode 1 (block): xu = acc*ratio*std + mean; t<L: residual -= xu;
//                 else d_out[b, t-L, c] += xu
__global__ __launch_bounds__(128) void k_irfft_gemm(
    const u16* __restrict__ Ar, const u16* __restrict__ Ai, int lda, int K,
    const u16* __restrict__ BcT, const u16* __restrict__ BsT, int ldb,
    const float* __restrict__ mean, const float* __restrict__ stdv,
    float* residual, float* outTotal, float ratio, int mode) {
  __shared__ u16 ar_sm[2][2048], ai_sm[2][2048];
  __shared__ u16 bc_sm[2][2048], bs_sm[2][2048];
  const int tid = threadIdx.x;
  const int row0 = blockIdx.y * 64, col0 = blockIdx.x * 64;
  const int wv = tid >> 5, ln = tid & 31, m16 = ln & 15, hf = ln >> 4;
  v8f acc[4];
#pragma unroll
  for (int j = 0; j < 4; ++j) acc[j] = vzero();
  stage(ar_sm[0], Ar, lda, row0, 0, tid);
  stage(ai_sm[0], Ai, lda, row0, 0, tid);
  stage(bc_sm[0], BcT, ldb, col0, 0, tid);
  stage(bs_sm[0], BsT, ldb, col0, 0, tid);
  const int nk = K >> 5;
  for (int ki = 0; ki < nk; ++ki) {
    const int cur = ki & 1;
    async_wait_all();
    __syncthreads();
    if (ki + 1 < nk) {
      int k1 = (ki + 1) << 5;
      stage(ar_sm[1 - cur], Ar, lda, row0, k1, tid);
      stage(ai_sm[1 - cur], Ai, lda, row0, k1, tid);
      stage(bc_sm[1 - cur], BcT, ldb, col0, k1, tid);
      stage(bs_sm[1 - cur], BsT, ldb, col0, k1, tid);
    }
    v16bf arf = ldsA(ar_sm[cur], wv * 16 + m16, hf);
    v16bf aif = ldsA(ai_sm[cur], wv * 16 + m16, hf);
    v16bf bcf[4], bsf[4];
#pragma unroll
    for (int j = 0; j < 4; ++j) {
      bcf[j] = ldsB(bc_sm[cur], j * 16 + m16, hf);
      bsf[j] = ldsB(bs_sm[cur], j * 16 + m16, hf);
    }
#pragma unroll
    for (int j = 0; j < 4; ++j) {
      acc[j] = wmma_bf16(arf, bcf[j], acc[j]);
      acc[j] = wmma_bf16(aif, bsf[j], acc[j]);
    }
  }
  float mu8[8], sd8[8];
#pragma unroll
  for (int r = 0; r < 8; ++r) {
    int grow = row0 + wv * 16 + hf * 8 + r;
    mu8[r] = mean[grow];
    sd8[r] = stdv[grow] * ratio;
  }
#pragma unroll
  for (int j = 0; j < 4; ++j)
#pragma unroll
    for (int r = 0; r < 8; ++r) {
      int grow = row0 + wv * 16 + hf * 8 + r;
      int t = col0 + j * 16 + m16;
      float xu = acc[j][r] * sd8[r] + mu8[r];
      if (mode == 0) {
        residual[(size_t)grow * L_ + t] = xu;
      } else if (t < L_) {
        residual[(size_t)grow * L_ + t] -= xu;
      } else {
        int b = grow >> 8, c = grow & 255;
        outTotal[((size_t)b * P_ + (t - L_)) * C_ + c] += xu;
      }
    }
}

// ------------------------------ launch -------------------------------------

extern "C" void kernel_launch(void* const* d_in, const int* in_sizes, int n_in,
                              void* d_out, int out_size, void* d_ws, size_t ws_size,
                              hipStream_t stream) {
  const float* x    = (const float*)d_in[0];
  const float* bb   = (const float*)d_in[1];
  const float* g_w1 = (const float*)d_in[2];
  const float* g_b1 = (const float*)d_in[3];
  const float* g_w2 = (const float*)d_in[4];
  const float* g_b2 = (const float*)d_in[5];
  const float* bw1  = (const float*)d_in[6];   // [NB,PF,F] complex (float pairs)
  const float* bb1  = (const float*)d_in[7];   // [NB,PF]  complex
  const float* bw2  = (const float*)d_in[8];   // [NB,PF,PF] complex
  const float* bb2  = (const float*)d_in[9];   // [NB,PF]  complex
  float* out = (float*)d_out;
  (void)in_sizes; (void)n_in; (void)out_size; (void)ws_size;

  char* ws = (char*)d_ws;
  size_t off = 0;
  auto alloc = [&](size_t bytes) -> char* {
    char* p = ws + off;
    off += (bytes + 255) & ~(size_t)255;
    return p;
  };
  u16* basRC  = (u16*)alloc((size_t)PADF * 512 * 2);   // rfft B^T
  u16* basRS  = (u16*)alloc((size_t)PADF * 512 * 2);
  u16* bas5C  = (u16*)alloc((size_t)512 * PADF * 2);   // irfft512 B^T
  u16* bas5S  = (u16*)alloc((size_t)512 * PADF * 2);
  u16* bas10C = (u16*)alloc((size_t)LOUT * PADP * 2);  // irfft1024 B^T
  u16* bas10S = (u16*)alloc((size_t)LOUT * PADP * 2);
  const size_t W1SZ = (size_t)PADP * PADF;   // per-block elems (B^T)
  const size_t W2SZ = (size_t)PADP * PADP;
  u16* w1R = (u16*)alloc(NB_ * W1SZ * 2);
  u16* w1I = (u16*)alloc(NB_ * W1SZ * 2);
  u16* w1N = (u16*)alloc(NB_ * W1SZ * 2);
  u16* w2R = (u16*)alloc(NB_ * W2SZ * 2);
  u16* w2I = (u16*)alloc(NB_ * W2SZ * 2);
  u16* w2N = (u16*)alloc(NB_ * W2SZ * 2);
  float* meanB = (float*)alloc((size_t)N_ * 4);
  float* stdB  = (float*)alloc((size_t)N_ * 4);
  float* gatein = (float*)alloc((size_t)B_ * PADF * 4);
  float* hbuf   = (float*)alloc((size_t)B_ * PADF * 4);
  float* scores = (float*)alloc((size_t)B_ * E_ * 4);
  float* wbin   = (float*)alloc((size_t)B_ * PADF * 4);
  int*   idxb   = (int*)alloc((E_ + 1) * 4);
  float* resid  = (float*)alloc((size_t)N_ * L_ * 4);
  // xn + specR + specI contiguous; up2 aliases them (N*1152 = 2*N*576 halves)
  u16* xn    = (u16*)alloc((size_t)N_ * L_ * 2);
  u16* specR = (u16*)alloc((size_t)N_ * PADF * 2);
  u16* specI = (u16*)alloc((size_t)N_ * PADF * 2);
  u16* upR   = (u16*)alloc((size_t)N_ * PADP * 2);
  u16* upI   = (u16*)alloc((size_t)N_ * PADP * 2);
  u16* up2R  = xn;
  u16* up2I  = (u16*)((char*)xn + (size_t)N_ * PADP * 2);

  // ---- constant prep (cheap, re-done every call for determinism) ----
  k_basis_rfft <<<(PADF * 512 + 255) / 256, 256, 0, stream>>>(basRC, basRS);
  k_basis_ir512<<<(512 * PADF + 255) / 256, 256, 0, stream>>>(bas5C, bas5S);
  k_basis_ir1024<<<(LOUT * PADP + 255) / 256, 256, 0, stream>>>(bas10C, bas10S);
  k_pack_w1<<<(NB_ * (int)W1SZ + 255) / 256, 256, 0, stream>>>(bw1, w1R, w1I, w1N);
  k_pack_w2<<<(NB_ * (int)W2SZ + 255) / 256, 256, 0, stream>>>(bw2, w2R, w2I, w2N);
  k_zero<<<(int)((OUT_TOTAL + 255) / 256), 256, 0, stream>>>(out, OUT_TOTAL);

  // ---- MoE ----
  k_stats<<<N_, 256, 0, stream>>>(x, 1, xn, meanB, stdB);
  k_rfft_gemm<<<dim3(PADF / 64, N_ / 64), 128, 0, stream>>>(
      xn, L_, basRC, basRS, L_, L_, specR, specI, PADF);
  k_gate_absmean<<<(B_ * PADF + 255) / 256, 256, 0, stream>>>(specR, specI, gatein);
  k_gate_h<<<(B_ * PADF + 255) / 256, 256, 0, stream>>>(gatein, g_w1, g_b1, hbuf);
  k_gate_scores<<<1, 64, 0, stream>>>(hbuf, g_w2, g_b2, bb, scores,
                                      out + OUT_BOUNDS, out + OUT_WEIGHT, idxb);
  k_wbin<<<(B_ * PADF + 255) / 256, 256, 0, stream>>>(scores, idxb, wbin);
  k_scale_fx<<<(int)(((size_t)N_ * PADF + 255) / 256), 256, 0, stream>>>(
      specR, specI, wbin, upR, upI);        // fxw staged in up region
  k_irfft_gemm<<<dim3(L_ / 64, N_ / 64), 128, 0, stream>>>(
      upR, upI, PADF, PADF, bas5C, bas5S, PADF, meanB, stdB,
      resid, (float*)nullptr, 1.0f, 0);

  // ---- 3 spectral blocks ----
  for (int i = 0; i < NB_; ++i) {
    k_stats<<<N_, 256, 0, stream>>>(resid, 0, xn, meanB, stdB);
    k_rfft_gemm<<<dim3(PADF / 64, N_ / 64), 128, 0, stream>>>(
        xn, L_, basRC, basRS, L_, L_, specR, specI, PADF);
    k_cgemm<<<dim3(PADP / 64, N_ / 64), 128, 0, stream>>>(
        specR, specI, PADF, PADF,
        w1R + (size_t)i * W1SZ, w1I + (size_t)i * W1SZ, w1N + (size_t)i * W1SZ,
        PADF, upR, upI, PADP, bb1 + (size_t)i * PF_ * 2, PF_, 1);
    k_cgemm<<<dim3(PADP / 64, N_ / 64), 128, 0, stream>>>(
        upR, upI, PADP, PADP,
        w2R + (size_t)i * W2SZ, w2I + (size_t)i * W2SZ, w2N + (size_t)i * W2SZ,
        PADP, up2R, up2I, PADP, bb2 + (size_t)i * PF_ * 2, PF_, 0);
    k_irfft_gemm<<<dim3(LOUT / 64, N_ / 64), 128, 0, stream>>>(
        up2R, up2I, PADP, PADP, bas10C, bas10S, PADP, meanB, stdB,
        resid, out, 2.0f /*LEN_RATIO*/, 1);
  }
}
